// EdgePoolLayer_58179626992082
// MI455X (gfx1250) — matrix operand order repaired and checked by reading, not compiled, not run on hardware
//
#include <hip/hip_runtime.h>
#include <hip/hip_bf16.h>

#define NN 50000
#define NE 400000
#define CF 256
#define NG 64
#define SORT_BLOCK 1024
#define MATCH_ROUNDS 48
#define CEILDIV(a,b) (((a)+(b)-1)/(b))

typedef _Float16 v16h __attribute__((ext_vector_type(16)));
typedef float    v8f  __attribute__((ext_vector_type(8)));

// ---------------- WMMA node projection: sA = x@w[:256], sB = x@w[256:] ---------
__global__ void k_node_dots(const float* __restrict__ x, const float* __restrict__ w,
                            float* __restrict__ sA, float* __restrict__ sB) {
  const int lane = threadIdx.x & 31;
  const int wave = threadIdx.x >> 5;
  const int tile = blockIdx.x * 8 + wave;          // 16 nodes per tile, NN/16 = 3125 exact
  if (tile >= NN / 16) return;
  const int m  = lane & 15;
  const int hi = lane >> 4;
  const float* xrow = x + (size_t)(tile * 16 + m) * CF;
  v8f acc = {};
  for (int kblk = 0; kblk < CF; kblk += 32) {
    // A (16x32 f16): slots 0..7 -> k = kblk+8*hi .. +7 ; slots 8..15 -> k = kblk+16+8*hi .. +7
    v16h a;
    const float* pa0 = xrow + kblk + 8 * hi;
    const float* pa1 = xrow + kblk + 16 + 8 * hi;
#pragma unroll
    for (int i = 0; i < 8; ++i) { a[i] = (_Float16)pa0[i]; a[8 + i] = (_Float16)pa1[i]; }
    // B (32x16 f16): lane n = m, slots 0..7 -> k = kblk+8*hi .. +7 ; slots 8..15 -> k = kblk+16+8*hi .. +7. Columns 0/1 = weight chunks.
    v16h bm;
    if (m < 2) {
      const float* wp = w + m * CF + kblk + 8 * hi;
#pragma unroll
      for (int i = 0; i < 8; ++i) { bm[i] = (_Float16)wp[i]; bm[8 + i] = (_Float16)wp[16 + i]; }
    } else {
#pragma unroll
      for (int i = 0; i < 16; ++i) bm[i] = (_Float16)0.0f;
    }
    acc = __builtin_amdgcn_wmma_f32_16x16x32_f16(false, a, false, bm, (short)0, acc, false, false);
  }
  // D (16x16 f32): lane n = m, vgpr v -> row v + 8*hi. Column 0 -> sA, column 1 -> sB.
  float* dstp = (m == 0) ? sA : (m == 1) ? sB : nullptr;
  if (dstp) {
    int base = tile * 16 + 8 * hi;
#pragma unroll
    for (int v = 0; v < 8; ++v) dstp[base + v] = acc[v];
  }
}

// ---------------- generic fill --------------------------------------------------
__global__ void k_fill32(unsigned int* p, unsigned int v, int n) {
  int i = blockIdx.x * 256 + threadIdx.x;
  if (i < n) p[i] = v;
}

// ---------------- edge softmax -------------------------------------------------
__device__ __forceinline__ unsigned int enc_f32(float f) {
  unsigned int b = __float_as_uint(f);
  return b ^ ((unsigned int)(((int)b) >> 31) | 0x80000000u);
}
__device__ __forceinline__ float dec_f32(unsigned int e) {
  unsigned int b = (e & 0x80000000u) ? (e ^ 0x80000000u) : ~e;
  return __uint_as_float(b);
}

__global__ void k_raw(const int* __restrict__ src, const int* __restrict__ dst,
                      const float* __restrict__ sA, const float* __restrict__ sB,
                      const float* __restrict__ b, float* __restrict__ raw,
                      unsigned int* __restrict__ mx) {
  int e = blockIdx.x * 256 + threadIdx.x;
  if (e >= NE) return;
  float r = sA[src[e]] + sB[dst[e]] + b[0];
  raw[e] = r;
  atomicMax(&mx[dst[e]], enc_f32(r));
}

__global__ void k_exp(const int* __restrict__ dst, const float* __restrict__ raw,
                      const unsigned int* __restrict__ mx, float* __restrict__ ex,
                      float* __restrict__ den) {
  int e = blockIdx.x * 256 + threadIdx.x;
  if (e >= NE) return;
  float v = __expf(raw[e] - dec_f32(mx[dst[e]]));
  ex[e] = v;
  atomicAdd(&den[dst[e]], v);
}

__global__ void k_score_keys(const int* __restrict__ dst, const float* __restrict__ ex,
                             const float* __restrict__ den, float* __restrict__ score,
                             unsigned long long* __restrict__ keys) {
  int e = blockIdx.x * 256 + threadIdx.x;
  if (e >= NE) return;
  float s = ex[e] / den[dst[e]] + 0.5f;
  score[e] = s;
  unsigned int desc = ~enc_f32(s);   // ascending sort of desc-key == descending score, tie: edge id
  keys[e] = ((unsigned long long)desc << 32) | (unsigned int)e;
}

// ---------------- LSD radix sort (u64 keys, payload in low bits) ----------------
__global__ void k_hist(const unsigned long long* __restrict__ keys, int n, int shift,
                       unsigned int* __restrict__ hist, int numBlocks) {
  __shared__ unsigned int lh[256];
  int t = threadIdx.x;
  lh[t] = 0;
  __syncthreads();
  int base = blockIdx.x * SORT_BLOCK;
  for (int i = t; i < SORT_BLOCK; i += 256) {
    int idx = base + i;
    if (idx < n) atomicAdd(&lh[(unsigned int)((keys[idx] >> shift) & 0xFF)], 1u);
  }
  __syncthreads();
  hist[t * numBlocks + blockIdx.x] = lh[t];
}

// single-block chunked exclusive scan; also returns total if requested
__global__ void k_scan(unsigned int* __restrict__ data, int n, unsigned int* total) {
  __shared__ unsigned int sums[256];
  int t = threadIdx.x;
  int chunk = (n + 255) / 256;
  int s = t * chunk;
  int e = s + chunk; if (e > n) e = n;
  unsigned int acc = 0;
  for (int i = s; i < e; ++i) acc += data[i];
  sums[t] = acc;
  __syncthreads();
  if (t == 0) {
    unsigned int run = 0;
    for (int i = 0; i < 256; ++i) { unsigned int v = sums[i]; sums[i] = run; run += v; }
    if (total) *total = run;
  }
  __syncthreads();
  unsigned int run = sums[t];
  for (int i = s; i < e; ++i) { unsigned int v = data[i]; data[i] = run; run += v; }
}

__global__ void k_scatter(const unsigned long long* __restrict__ in,
                          unsigned long long* __restrict__ out, int n, int shift,
                          const unsigned int* __restrict__ hist, int numBlocks) {
  __shared__ unsigned long long elems[SORT_BLOCK];
  __shared__ unsigned int pos[SORT_BLOCK];
  __shared__ unsigned int cnt[256];
  int t = threadIdx.x;
  int base = blockIdx.x * SORT_BLOCK;
  int m = n - base; if (m > SORT_BLOCK) m = SORT_BLOCK;
  for (int i = t; i < m; i += 256) elems[i] = in[base + i];
  cnt[t] = hist[t * numBlocks + blockIdx.x];
  __syncthreads();
  if (t == 0) {  // stable within block
    for (int i = 0; i < m; ++i) {
      unsigned int d = (unsigned int)((elems[i] >> shift) & 0xFF);
      pos[i] = cnt[d]++;
    }
  }
  __syncthreads();
  for (int i = t; i < m; i += 256) out[pos[i]] = elems[i];
}

// ---------------- parallel greedy matching (locally-dominant == sequential greedy)
__global__ void k_pri(const unsigned long long* __restrict__ keys, unsigned int* __restrict__ pri) {
  int j = blockIdx.x * 256 + threadIdx.x;
  if (j < NE) pri[(unsigned int)keys[j]] = (unsigned int)j;
}
__global__ void k_bid(const int* __restrict__ src, const int* __restrict__ dst,
                      const unsigned int* __restrict__ pri, const unsigned int* __restrict__ state,
                      unsigned int* __restrict__ best) {
  int e = blockIdx.x * 256 + threadIdx.x;
  if (e >= NE || state[e] != 0) return;
  unsigned int p = pri[e];
  atomicMin(&best[src[e]], p);
  atomicMin(&best[dst[e]], p);
}
__global__ void k_choose(const int* __restrict__ src, const int* __restrict__ dst,
                         const unsigned int* __restrict__ pri, const unsigned int* __restrict__ best,
                         unsigned int* __restrict__ state, unsigned int* __restrict__ matched) {
  int e = blockIdx.x * 256 + threadIdx.x;
  if (e >= NE || state[e] != 0) return;
  int u = src[e], v = dst[e];
  unsigned int p = pri[e];
  if (best[u] == p && best[v] == p) { state[e] = 1; matched[u] = 1; matched[v] = 1; }
}
__global__ void k_kill(const int* __restrict__ src, const int* __restrict__ dst,
                       const unsigned int* __restrict__ matched, unsigned int* __restrict__ state) {
  int e = blockIdx.x * 256 + threadIdx.x;
  if (e >= NE || state[e] != 0) return;
  if (matched[src[e]] | matched[dst[e]]) state[e] = 2;
}

// ---------------- cluster ids ---------------------------------------------------
__global__ void k_cs(const unsigned long long* __restrict__ keys,
                     const unsigned int* __restrict__ state, unsigned int* __restrict__ cs) {
  int j = blockIdx.x * 256 + threadIdx.x;
  if (j < NE) cs[j] = (state[(unsigned int)keys[j]] == 1u) ? 1u : 0u;
}
__global__ void k_assign(const unsigned long long* __restrict__ keys,
                         const unsigned int* __restrict__ state, const unsigned int* __restrict__ cs,
                         const int* __restrict__ src, const int* __restrict__ dst,
                         const float* __restrict__ score, unsigned int* __restrict__ mapping,
                         float* __restrict__ node_scores) {
  int j = blockIdx.x * 256 + threadIdx.x;
  if (j >= NE) return;
  unsigned int e = (unsigned int)keys[j];
  if (state[e] == 1u) {
    unsigned int id = cs[j];               // exclusive count of earlier chosen == cnt at pick time
    mapping[src[e]] = id;
    mapping[dst[e]] = id;
    node_scores[id] = score[e];
  }
}
__global__ void k_rem(const unsigned int* __restrict__ matched, unsigned int* __restrict__ rem) {
  int v = blockIdx.x * 256 + threadIdx.x;
  if (v < NN) rem[v] = matched[v] ? 0u : 1u;
}
__global__ void k_map_rem(const unsigned int* __restrict__ matched, const unsigned int* __restrict__ rem,
                          const unsigned int* __restrict__ cntTotal, unsigned int* __restrict__ mapping) {
  int v = blockIdx.x * 256 + threadIdx.x;
  if (v >= NN) return;
  if (!matched[v]) mapping[v] = *cntTotal + rem[v];
}
__global__ void k_nnew(const unsigned int* __restrict__ cntTotal, const unsigned int* __restrict__ remTotal,
                       unsigned int* __restrict__ nnew_ws, int* __restrict__ out_nnew) {
  unsigned int nn = *cntTotal + *remTotal;
  *nnew_ws = nn;
  *out_nnew = (int)nn;
}

// ---------------- contraction ---------------------------------------------------
__global__ void k_invert(const unsigned int* __restrict__ mapping, unsigned int* __restrict__ mcnt,
                         unsigned int* __restrict__ members) {
  int v = blockIdx.x * 256 + threadIdx.x;
  if (v >= NN) return;
  unsigned int c = mapping[v];
  unsigned int i = atomicAdd(&mcnt[c], 1u);
  if (i < 2) members[c * 2 + i] = (unsigned int)v;
}
__global__ void k_newx(const float* __restrict__ x, const unsigned int* __restrict__ mcnt,
                       const unsigned int* __restrict__ members, const float* __restrict__ node_scores,
                       float* __restrict__ new_x) {
  int idx = blockIdx.x * 256 + threadIdx.x;         // row*64 + c4
  int row = idx >> 6;
  if (row >= NN) return;
  int c = (idx & 63) << 2;
  unsigned int cnt = mcnt[row];
  float a0 = 0.f, a1 = 0.f, a2 = 0.f, a3 = 0.f;
  if (cnt >= 1) {
    const float* p = x + (size_t)members[row * 2] * CF + c;
    if (cnt >= 2) __builtin_prefetch(x + (size_t)members[row * 2 + 1] * CF + c, 0, 0);
    a0 = p[0]; a1 = p[1]; a2 = p[2]; a3 = p[3];
  }
  if (cnt >= 2) {
    const float* p = x + (size_t)members[row * 2 + 1] * CF + c;
    a0 += p[0]; a1 += p[1]; a2 += p[2]; a3 += p[3];
  }
  float s = node_scores[row];
  float* o = new_x + (size_t)row * CF + c;
  o[0] = a0 * s; o[1] = a1 * s; o[2] = a2 * s; o[3] = a3 * s;
}

// ---------------- new edges / batch --------------------------------------------
__global__ void k_nekeys(const int* __restrict__ src, const int* __restrict__ dst,
                         const unsigned int* __restrict__ mapping, unsigned long long* __restrict__ keys) {
  int e = blockIdx.x * 256 + threadIdx.x;
  if (e >= NE) return;
  unsigned int k = mapping[src[e]] * (unsigned int)NN + mapping[dst[e]];  // < 2^32
  keys[e] = ((unsigned long long)k << 32) | (unsigned int)e;
}
__global__ void k_wedges(const unsigned long long* __restrict__ keys, const unsigned int* __restrict__ mapping,
                         const int* __restrict__ src, const int* __restrict__ dst, int* __restrict__ out_edges) {
  int j = blockIdx.x * 256 + threadIdx.x;
  if (j >= NE) return;
  unsigned int e = (unsigned int)keys[j];
  out_edges[j]      = (int)mapping[src[e]];
  out_edges[NE + j] = (int)mapping[dst[e]];
}
__global__ void k_nbmax(const unsigned int* __restrict__ mapping, const int* __restrict__ batch,
                        int* __restrict__ nb) {
  int v = blockIdx.x * 256 + threadIdx.x;
  if (v >= NN) return;
  atomicMax(&nb[mapping[v]], batch[v]);
}
__global__ void k_nbwrite(const int* __restrict__ nb, const unsigned int* __restrict__ nnew,
                          int* __restrict__ out_batch) {
  int i = blockIdx.x * 256 + threadIdx.x;
  if (i >= NN) return;
  out_batch[i] = (i < (int)*nnew) ? nb[i] : 0;
}

// ---------------- host orchestration -------------------------------------------
extern "C" void kernel_launch(void* const* d_in, const int* in_sizes, int n_in,
                              void* d_out, int out_size, void* d_ws, size_t ws_size,
                              hipStream_t stream) {
  const float* x    = (const float*)d_in[0];
  const int*   src  = (const int*)d_in[1];
  const int*   dst  = src + NE;
  const int*   batch= (const int*)d_in[2];
  const float* w    = (const float*)d_in[3];
  const float* b    = (const float*)d_in[4];

  // output layout: new_x | new_edge_index | new_batch | n_new | node_scores
  float* out_newx   = (float*)d_out;
  int*   out_edges  = (int*)(out_newx + (size_t)NN * CF);
  int*   out_batch  = out_edges + 2 * NE;
  int*   out_nnew   = out_batch + NN;
  float* out_scores = (float*)(out_nnew + 1);

  // workspace carve-up
  char* p = (char*)d_ws;
  auto alloc = [&](size_t bytes) { char* r = p; p += (bytes + 255) & ~(size_t)255; return r; };
  float* sA   = (float*)alloc(NN * 4);
  float* sB   = (float*)alloc(NN * 4);
  float* raw  = (float*)alloc(NE * 4);
  float* ex   = (float*)alloc(NE * 4);
  float* score= (float*)alloc(NE * 4);
  unsigned int* mx  = (unsigned int*)alloc(NN * 4);
  float* den  = (float*)alloc(NN * 4);
  unsigned long long* keyA = (unsigned long long*)alloc(NE * 8);
  unsigned long long* keyB = (unsigned long long*)alloc(NE * 8);
  const int numSortBlocks = CEILDIV(NE, SORT_BLOCK);
  unsigned int* hist = (unsigned int*)alloc((size_t)256 * numSortBlocks * 4);
  unsigned int* pri    = (unsigned int*)alloc(NE * 4);
  unsigned int* state  = (unsigned int*)alloc(NE * 4);
  unsigned int* best   = (unsigned int*)alloc(NN * 4);
  unsigned int* matched= (unsigned int*)alloc(NN * 4);
  unsigned int* cs     = (unsigned int*)alloc(NE * 4);
  unsigned int* mapping= (unsigned int*)alloc(NN * 4);
  unsigned int* rem    = (unsigned int*)alloc(NN * 4);
  unsigned int* mcnt   = (unsigned int*)alloc(NN * 4);
  unsigned int* members= (unsigned int*)alloc((size_t)2 * NN * 4);
  int* nb              = (int*)alloc(NN * 4);
  unsigned int* cntTotal = (unsigned int*)alloc(4);
  unsigned int* remTotal = (unsigned int*)alloc(4);
  unsigned int* nnew_ws  = (unsigned int*)alloc(4);

  const int gN = CEILDIV(NN, 256), gE = CEILDIV(NE, 256);

  auto radix_sort = [&](unsigned long long* a, unsigned long long* bb) {
    for (int pass = 0; pass < 8; ++pass) {
      int shift = pass * 8;
      unsigned long long* s = (pass & 1) ? bb : a;
      unsigned long long* d = (pass & 1) ? a : bb;
      k_hist<<<numSortBlocks, 256, 0, stream>>>(s, NE, shift, hist, numSortBlocks);
      k_scan<<<1, 256, 0, stream>>>(hist, 256 * numSortBlocks, nullptr);
      k_scatter<<<numSortBlocks, 256, 0, stream>>>(s, d, NE, shift, hist, numSortBlocks);
    }
  }; // 8 passes -> sorted result back in `a`

  // 1) node projections via WMMA
  k_node_dots<<<CEILDIV(NN / 16, 8), 256, 0, stream>>>(x, w, sA, sB);

  // 2) per-dst softmax -> score + sort keys
  k_fill32<<<gN, 256, 0, stream>>>(mx, 0u, NN);
  k_fill32<<<gN, 256, 0, stream>>>((unsigned int*)den, 0u, NN);
  k_raw<<<gE, 256, 0, stream>>>(src, dst, sA, sB, b, raw, mx);
  k_exp<<<gE, 256, 0, stream>>>(dst, raw, mx, ex, den);
  k_score_keys<<<gE, 256, 0, stream>>>(dst, ex, den, score, keyA);

  // 3) sort edges by descending score (stable)
  radix_sort(keyA, keyB);
  k_pri<<<gE, 256, 0, stream>>>(keyA, pri);

  // 4) greedy matching == iterated locally-dominant matching
  k_fill32<<<gE, 256, 0, stream>>>(state, 0u, NE);
  k_fill32<<<gN, 256, 0, stream>>>(matched, 0u, NN);
  k_fill32<<<gN, 256, 0, stream>>>(mapping, 0u, NN);
  for (int r = 0; r < MATCH_ROUNDS; ++r) {
    k_fill32<<<gN, 256, 0, stream>>>(best, 0xFFFFFFFFu, NN);
    k_bid<<<gE, 256, 0, stream>>>(src, dst, pri, state, best);
    k_choose<<<gE, 256, 0, stream>>>(src, dst, pri, best, state, matched);
    k_kill<<<gE, 256, 0, stream>>>(src, dst, matched, state);
  }

  // 5) cluster ids in pick order + node scores + remaining nodes
  k_cs<<<gE, 256, 0, stream>>>(keyA, state, cs);
  k_scan<<<1, 256, 0, stream>>>(cs, NE, cntTotal);
  k_fill32<<<gN, 256, 0, stream>>>((unsigned int*)out_scores, 0x3F800000u, NN);  // 1.0f
  k_assign<<<gE, 256, 0, stream>>>(keyA, state, cs, src, dst, score, mapping, out_scores);
  k_rem<<<gN, 256, 0, stream>>>(matched, rem);
  k_scan<<<1, 256, 0, stream>>>(rem, NN, remTotal);
  k_map_rem<<<gN, 256, 0, stream>>>(matched, rem, cntTotal, mapping);
  k_nnew<<<1, 1, 0, stream>>>(cntTotal, remTotal, nnew_ws, out_nnew);

  // 6) new_x: invert mapping (clusters have <=2 members), gather + scale
  k_fill32<<<gN, 256, 0, stream>>>(mcnt, 0u, NN);
  k_invert<<<gN, 256, 0, stream>>>(mapping, mcnt, members);
  k_newx<<<CEILDIV(NN * 64, 256), 256, 0, stream>>>(x, mcnt, members, out_scores, out_newx);

  // 7) new_edge_index: remap + stable sort by ne0*n+ne1
  k_nekeys<<<gE, 256, 0, stream>>>(src, dst, mapping, keyA);
  radix_sort(keyA, keyB);
  k_wedges<<<gE, 256, 0, stream>>>(keyA, mapping, src, dst, out_edges);

  // 8) new_batch: segment_max(batch, mapping), masked by n_new
  k_fill32<<<gN, 256, 0, stream>>>((unsigned int*)nb, 0x80000000u, NN);  // INT_MIN
  k_nbmax<<<gN, 256, 0, stream>>>(mapping, batch, nb);
  k_nbwrite<<<gN, 256, 0, stream>>>(nb, nnew_ws, out_batch);
}
